// CustomLSTMCell_9517647528292
// MI455X (gfx1250) — matrix operand order repaired and checked by reading, not compiled, not run on hardware
//
#include <hip/hip_runtime.h>

typedef __attribute__((ext_vector_type(16))) __bf16 v16bf;
typedef __attribute__((ext_vector_type(8)))  float  v8f;

#define B_ROWS 8192
#define HID    1024
#define KDIM   2048   // IN_DIM + HID
#define NGATE  4

#define TM     128    // rows per block
#define TN     32     // cols per gate per block
#define KSTEP  64     // k per pipeline stage (2 WMMA sub-steps of 32)
#define NSTAGE (KDIM / KSTEP)   // 32
#define LDA    72     // padded LDS stride (bf16 units) -> 144B rows, b128-aligned
#define LDB    72
#define ABUF_BYTES (TM * LDA * 2)          // 18432
#define BBUF_BYTES (NGATE * TN * LDB * 2)  // 18432

static __device__ __forceinline__ unsigned short f2bf(float f) {
    unsigned int u = __float_as_uint(f);
    u += 0x7FFFu + ((u >> 16) & 1u);           // round-to-nearest-even
    return (unsigned short)(u >> 16);
}
static __device__ __forceinline__ float sigmoidf_(float x) {
    return 1.0f / (1.0f + __expf(-x));
}
static __device__ __forceinline__ float tanhf_(float x) {
    float e = __expf(-2.0f * x);
    return (1.0f - e) / (1.0f + e);
}

// gfx1250 async global->LDS copy, 16B per lane, tracked with ASYNCcnt.
static __device__ __forceinline__ void async_b128(unsigned lds_off, const void* gaddr) {
    asm volatile("global_load_async_to_lds_b128 %0, %1, off"
                 :: "v"(lds_off), "v"((unsigned long long)(uintptr_t)gaddr)
                 : "memory");
}
#define S_WAIT_ASYNC(n) asm volatile("s_wait_asynccnt " #n ::: "memory")

// ---- kernel 1: combined = [Z, h_prev] in bf16 ------------------------------
__global__ void build_A(const float* __restrict__ Z, const float* __restrict__ Hp,
                        unsigned short* __restrict__ A) {
    unsigned int idx  = blockIdx.x * blockDim.x + threadIdx.x; // 4 elems/thread
    unsigned int base = idx * 4u;
    unsigned int r    = base >> 11;      // /2048
    unsigned int k    = base & 2047u;
    const float* src = (k < 1024u) ? (Z + (size_t)r * 1024u + k)
                                   : (Hp + (size_t)r * 1024u + (k - 1024u));
    float4 v = *(const float4*)src;
    ushort4 o;
    o.x = f2bf(v.x); o.y = f2bf(v.y); o.z = f2bf(v.z); o.w = f2bf(v.w);
    *(ushort4*)(A + base) = o;
}

// ---- kernel 2: Wt[g][n][k] = bf16(W_g[k][n])  (LDS tiled transpose) --------
__global__ void transpose_W(const float* __restrict__ Wi, const float* __restrict__ Wf,
                            const float* __restrict__ Wo, const float* __restrict__ Wg,
                            unsigned short* __restrict__ Wt) {
    __shared__ unsigned short tile[32][33];
    const float* Wsrc[4] = {Wi, Wf, Wo, Wg};
    const unsigned int g   = blockIdx.z;
    const unsigned int k0  = blockIdx.y * 32;
    const unsigned int n0  = blockIdx.x * 32;
    const float* src = Wsrc[g];
    const unsigned int tid = threadIdx.x;
    const unsigned int rlo = tid >> 5;   // 0..7
    const unsigned int c   = tid & 31u;
#pragma unroll
    for (int i = 0; i < 4; ++i) {
        unsigned int kk = rlo + i * 8;
        tile[kk][c] = f2bf(src[(size_t)(k0 + kk) * HID + n0 + c]);
    }
    __syncthreads();
    unsigned short* dst = Wt + (size_t)g * HID * KDIM;
#pragma unroll
    for (int i = 0; i < 4; ++i) {
        unsigned int nn = rlo + i * 8;
        dst[(size_t)(n0 + nn) * KDIM + k0 + c] = tile[c][nn];
    }
}

// ---- kernel 3: fused 4-gate GEMM (WMMA bf16) + LSTM epilogue ---------------
// Async double-buffered pipeline: stage = 128x64 A tile + 4x(32x64) B tiles.
static __device__ __forceinline__ void issue_stage(
    const unsigned short* __restrict__ A, const unsigned short* __restrict__ Wt,
    unsigned rowBase, unsigned n0, unsigned k0,
    unsigned ldsAoff, unsigned ldsBoff, unsigned tid)
{
#pragma unroll
    for (int it = 0; it < 4; ++it) {
        unsigned slot = tid + (unsigned)it * 256u;
        unsigned row  = slot >> 3;      // 0..127
        unsigned seg  = slot & 7u;      // 16B segment within 64-elem strip
        async_b128(ldsAoff + (row * LDA + seg * 8) * 2,
                   A + (size_t)(rowBase + row) * KDIM + k0 + seg * 8);
        unsigned g = row >> 5, n = row & 31u;
        async_b128(ldsBoff + (row * LDB + seg * 8) * 2,
                   Wt + ((size_t)g * HID + n0 + n) * KDIM + k0 + seg * 8);
    }
}

__global__ __launch_bounds__(256)
void lstm_wmma(const unsigned short* __restrict__ A,
               const unsigned short* __restrict__ Wt,
               const float* __restrict__ bi, const float* __restrict__ bf,
               const float* __restrict__ bo, const float* __restrict__ bg,
               const float* __restrict__ Cprev,
               float* __restrict__ Hout, float* __restrict__ Cout)
{
    __shared__ unsigned short ldsA[2][TM * LDA];
    __shared__ unsigned short ldsB[2][NGATE * TN * LDB];

    const unsigned int tid  = threadIdx.x;
    const unsigned int lane = tid & 31u;
    const unsigned int wv   = tid >> 5;     // 0..7 (wave32)
    const unsigned int half = lane >> 4;    // K-half selector per ISA layout
    const unsigned int col  = lane & 15u;   // M (for A) / N (for B,C,D)

    const unsigned int rowBase = blockIdx.y * TM;
    const unsigned int n0      = blockIdx.x * TN;

    const unsigned int aBase = (unsigned int)(uintptr_t)(&ldsA[0][0]);
    const unsigned int bBase = (unsigned int)(uintptr_t)(&ldsB[0][0]);

    v8f acc[NGATE][2];
#pragma unroll
    for (int g = 0; g < NGATE; ++g)
#pragma unroll
        for (int t = 0; t < 2; ++t)
#pragma unroll
            for (int e = 0; e < 8; ++e) acc[g][t][e] = 0.0f;

    issue_stage(A, Wt, rowBase, n0, 0u, aBase, bBase, tid);   // prologue: 8 async ops

    for (int st = 0; st < NSTAGE; ++st) {
        const int buf = st & 1;
        if (st + 1 < NSTAGE) {
            issue_stage(A, Wt, rowBase, n0, (unsigned)(st + 1) * KSTEP,
                        aBase + (unsigned)(buf ^ 1) * ABUF_BYTES,
                        bBase + (unsigned)(buf ^ 1) * BBUF_BYTES, tid);
            S_WAIT_ASYNC(0x8);   // the 8 older ops (current stage) are done
        } else {
            S_WAIT_ASYNC(0x0);
        }
        __syncthreads();         // all waves' deposits for this stage visible

#pragma unroll
        for (int s = 0; s < 2; ++s) {   // two K=32 sub-steps per stage
            // A fragment: 16-bit A 16x32 layout (dword groups half*4, 8+half*4)
            union { uint4 q[2]; v16bf v; } aF;
            {
                const unsigned short* pr = &ldsA[buf][(wv * 16 + col) * LDA + s * 32];
                aF.q[0] = *(const uint4*)(pr + half * 8);
                aF.q[1] = *(const uint4*)(pr + 16 + half * 8);
            }
#pragma unroll
            for (int g = 0; g < NGATE; ++g) {
#pragma unroll
                for (int t = 0; t < 2; ++t) {
                    // B fragment: 32x16, lane=N; lanes0-15 K=0..7,16..23, lanes16-31 K=8..15,24..31
                    union { uint4 q[2]; v16bf v; } bFr;
                    const unsigned short* pb =
                        &ldsB[buf][(g * TN + t * 16 + col) * LDB + s * 32 + half * 8];
                    bFr.q[0] = *(const uint4*)(pb);
                    bFr.q[1] = *(const uint4*)(pb + 16);
                    acc[g][t] = __builtin_amdgcn_wmma_f32_16x16x32_bf16(
                        false, aF.v, false, bFr.v, (short)0, acc[g][t], false, false);
                }
            }
        }
        __syncthreads();         // safe to overwrite buf at next issue
    }

    // epilogue: C/D layout -> lane: N = col, M = j + 8*half within 16-row strip
    const unsigned int mRow = rowBase + wv * 16 + half * 8;
#pragma unroll
    for (int t = 0; t < 2; ++t) {
        unsigned int n = n0 + t * 16 + col;
        float vbi = bi[n], vbf = bf[n], vbo = bo[n], vbg = bg[n];
#pragma unroll
        for (int j = 0; j < 8; ++j) {
            unsigned int m = mRow + j;
            float xi = sigmoidf_(acc[0][t][j] + vbi);
            float xf = sigmoidf_(acc[1][t][j] + vbf);
            float xo = sigmoidf_(acc[2][t][j] + vbo);
            float xg = tanhf_  (acc[3][t][j] + vbg);
            float cp = Cprev[(size_t)m * HID + n];
            float cn = xf * cp + xi * xg;
            float hn = xo * tanhf_(cn);
            Hout[(size_t)m * HID + n] = hn;
            Cout[(size_t)m * HID + n] = cn;
        }
    }
}

extern "C" void kernel_launch(void* const* d_in, const int* in_sizes, int n_in,
                              void* d_out, int out_size, void* d_ws, size_t ws_size,
                              hipStream_t stream) {
    (void)in_sizes; (void)n_in; (void)out_size; (void)ws_size;
    const float* Z  = (const float*)d_in[0];
    const float* Hp = (const float*)d_in[1];
    const float* Cp = (const float*)d_in[2];
    const float* Wi = (const float*)d_in[3];
    const float* bi = (const float*)d_in[4];
    const float* Wf = (const float*)d_in[5];
    const float* bf = (const float*)d_in[6];
    const float* Wo = (const float*)d_in[7];
    const float* bo = (const float*)d_in[8];
    const float* Wg = (const float*)d_in[9];
    const float* bg = (const float*)d_in[10];

    unsigned short* Abf = (unsigned short*)d_ws;                  // 33.55 MB
    unsigned short* Wt  = Abf + (size_t)B_ROWS * KDIM;            // 16.78 MB

    float* Hout = (float*)d_out;
    float* Cout = Hout + (size_t)B_ROWS * HID;

    build_A<<<dim3((B_ROWS * KDIM / 4) / 256), dim3(256), 0, stream>>>(Z, Hp, Abf);
    transpose_W<<<dim3(HID / 32, KDIM / 32, 4), dim3(256), 0, stream>>>(Wi, Wf, Wo, Wg, Wt);
    lstm_wmma<<<dim3(HID / TN, B_ROWS / TM), dim3(256), 0, stream>>>(
        Abf, Wt, bi, bf, bo, bg, Cp, Hout, Cout);
}
